// HardResetMiniJMamba_15418932592973
// MI455X (gfx1250) — matrix operand order repaired and checked by reading, not compiled, not run on hardware
//
#include <hip/hip_runtime.h>
#include <hip/hip_bf16.h>
#include <math.h>

// ---------------------------------------------------------------------------
// HardResetMiniJMamba forward for MI455X (gfx1250): f16 WMMA GEMMs with fused
// epilogues (pos-emb, GLU/silu, residual+periodic decay, masked heads).
// ---------------------------------------------------------------------------

typedef __attribute__((ext_vector_type(16))) _Float16 v16h;
typedef __attribute__((ext_vector_type(8)))  float    v8f;

#define BSROWS  16384   // B*S tokens
#define SEQ     2048
#define DMODEL  512
#define DINNER  1024
#define NFRAME  128
#define NVOCAB  256
#define NLAYERS 4

enum { EPI_INPROJ = 0, EPI_GLU = 1, EPI_OUT = 2, EPI_HEAD = 3 };

// LDS strides in halves; 40 halves = 80 B keeps every 8-half chunk 16B-aligned
#define ASTRIDE 40
#define BSTRIDE 40

// A fragment (16x32 f16, row-major tile in LDS):
// lanes 0-15 / 16-31 = halves h=0/1; elems 0..7 -> K = h*8+e, elems 8..15 ->
// K = 16 + h*8 + (e-8). Row = rowBase + (lane & 15).  (ISA 7.12.2)
__device__ __forceinline__ v16h wmma_ld_a(const _Float16* As, int rowBase, int lane) {
    const int h = lane >> 4;
    const int r = lane & 15;
    const _Float16* p = As + (size_t)(rowBase + r) * ASTRIDE;
    union { uint4 q[2]; v16h v; } u;
    u.q[0] = *(const uint4*)(p + h * 8);        // K = h*8 .. h*8+7
    u.q[1] = *(const uint4*)(p + 16 + h * 8);   // K = 16+h*8 .. 16+h*8+7
    return u.v;
}

// B fragment (32x16 f16). Bs is stored transposed: Bs[n][k]. Lane holds column
// n = lane&15; elems 0..7 -> K = h*8 + e, elems 8..15 -> K = 16 + h*8 + (e-8).
__device__ __forceinline__ v16h wmma_ld_b(const _Float16* Bs, int colBase, int lane) {
    const int h = lane >> 4;
    const int n = lane & 15;
    const _Float16* p = Bs + (size_t)(colBase + n) * BSTRIDE;
    union { uint4 q[2]; v16h v; } u;
    u.q[0] = *(const uint4*)(p + h * 8);
    u.q[1] = *(const uint4*)(p + 16 + h * 8);
    return u.v;
}

__device__ __forceinline__ v8f wmma32(v16h a, v16h b, v8f c) {
    return __builtin_amdgcn_wmma_f32_16x16x32_f16(false, a, false, b,
                                                  (short)0, c, false, false);
}

// ---------------------------------------------------------------------------
// Tiled GEMM: block tile 128(M) x 64(N), 8 waves (4x2), 32x32 per wave,
// K-step 32.  A: [M,K] f16 row-major.  B: [K,ldb] f16 row-major.
// ---------------------------------------------------------------------------
template <int EPI>
__global__ __launch_bounds__(256) void gemm_wmma_f16(
    const _Float16* __restrict__ A, const _Float16* __restrict__ B,
    const float* __restrict__ bias, float* __restrict__ f32out,
    _Float16* __restrict__ f16out, const float* __restrict__ pos,
    const unsigned char* __restrict__ mask,
    int K, int ldb, int outLd, int gOff)
{
    __shared__ _Float16 As[128 * ASTRIDE];
    __shared__ _Float16 Bs[64 * BSTRIDE];
    __shared__ _Float16 Bs2[64 * BSTRIDE];   // g-columns for GLU

    const int tid  = threadIdx.x;
    const int lane = tid & 31;
    const int wave = tid >> 5;
    const int wm   = wave >> 1;          // 0..3
    const int wn   = wave & 1;           // 0..1
    const int rowBase = blockIdx.x * 128;
    const int colBase = blockIdx.y * 64;

    v8f acc[2][2]  = {};
    v8f accG[2][2] = {};

    // cooperative staging assignments
    const int ar = tid >> 1;             // A tile row 0..127
    const int ac = (tid & 1) * 16;       // A col chunk (halves)
    const int bk = tid >> 3;             // B tile K row 0..31
    const int bn = (tid & 7) * 8;        // B tile col 0..56

    for (int k0 = 0; k0 < K; k0 += 32) {
        __syncthreads();
        // ---- stage A: 128x32 halves, two b128 loads/thread
        {
            const _Float16* g = A + (size_t)(rowBase + ar) * K + k0 + ac;
            uint4 q0 = *(const uint4*)(g);
            uint4 q1 = *(const uint4*)(g + 8);
            *(uint4*)(&As[ar * ASTRIDE + ac])     = q0;
            *(uint4*)(&As[ar * ASTRIDE + ac + 8]) = q1;
        }
        // ---- stage B transposed: Bs[n][k]
        {
            const _Float16* g = B + (size_t)(k0 + bk) * ldb + colBase + bn;
            union { uint4 q; _Float16 h[8]; } u;
            u.q = *(const uint4*)(g);
#pragma unroll
            for (int j = 0; j < 8; ++j)
                Bs[(bn + j) * BSTRIDE + bk] = u.h[j];
            if (EPI == EPI_GLU) {
                const _Float16* g2 = B + (size_t)(k0 + bk) * ldb + gOff + colBase + bn;
                union { uint4 q; _Float16 h[8]; } u2;
                u2.q = *(const uint4*)(g2);
#pragma unroll
                for (int j = 0; j < 8; ++j)
                    Bs2[(bn + j) * BSTRIDE + bk] = u2.h[j];
            }
        }
        // ---- prefetch next K tile (global_prefetch_b8)
        if (k0 + 32 < K) {
            __builtin_prefetch(A + (size_t)(rowBase + ar) * K + (k0 + 32) + ac, 0, 1);
            __builtin_prefetch(B + (size_t)(k0 + 32 + bk) * ldb + colBase + bn, 0, 1);
        }
        __syncthreads();

        // ---- matrix core: 4 (or 8) WMMAs per wave per K step
        v16h a0 = wmma_ld_a(As, wm * 32,      lane);
        v16h a1 = wmma_ld_a(As, wm * 32 + 16, lane);
        v16h b0 = wmma_ld_b(Bs, wn * 32,      lane);
        v16h b1 = wmma_ld_b(Bs, wn * 32 + 16, lane);
        acc[0][0] = wmma32(a0, b0, acc[0][0]);
        acc[0][1] = wmma32(a0, b1, acc[0][1]);
        acc[1][0] = wmma32(a1, b0, acc[1][0]);
        acc[1][1] = wmma32(a1, b1, acc[1][1]);
        if (EPI == EPI_GLU) {
            v16h c0 = wmma_ld_b(Bs2, wn * 32,      lane);
            v16h c1 = wmma_ld_b(Bs2, wn * 32 + 16, lane);
            accG[0][0] = wmma32(a0, c0, accG[0][0]);
            accG[0][1] = wmma32(a0, c1, accG[0][1]);
            accG[1][0] = wmma32(a1, c0, accG[1][0]);
            accG[1][1] = wmma32(a1, c1, accG[1][1]);
        }
    }

    // ---- epilogue.  D layout: VGPR r -> row = h*8 + r, col = lane & 15.
    const int h  = lane >> 4;
    const int nn = lane & 15;
#pragma unroll
    for (int sm = 0; sm < 2; ++sm)
#pragma unroll
    for (int sn = 0; sn < 2; ++sn)
#pragma unroll
    for (int r = 0; r < 8; ++r) {
        const int row = rowBase + wm * 32 + sm * 16 + h * 8 + r;
        const int col = colBase + wn * 32 + sn * 16 + nn;
        const float v = acc[sm][sn][r];
        if (EPI == EPI_INPROJ) {
            const int s = row & (SEQ - 1);      // pos index (SEQ < MF, no clip)
            f32out[(size_t)row * outLd + col] =
                v + bias[col] + pos[(size_t)s * outLd + col];
        } else if (EPI == EPI_GLU) {
            const float uu = v + bias[col];
            const float gg = accG[sm][sn][r] + bias[col + gOff];
            const float sg = gg / (1.0f + __expf(-gg));      // silu(g)
            f16out[(size_t)row * outLd + col] = (_Float16)(uu * sg);
        } else if (EPI == EPI_OUT) {
            const int s = row & (SEQ - 1);
            const float decay = (((s + 1) % 10) == 0) ? 0.1f : 1.0f;
            const size_t ix = (size_t)row * outLd + col;
            f32out[ix] = (f32out[ix] + (v + bias[col])) * decay;
        } else {                                // EPI_HEAD
            const float o = v + bias[col];
            f32out[(size_t)row * outLd + col] = mask[row] ? o : 0.0f;
        }
    }
}

// ---------------------------------------------------------------------------
// LayerNorm over D=512, one block (256 thr) per token, f16 output.
// ---------------------------------------------------------------------------
__global__ __launch_bounds__(256) void layernorm_f16(
    const float* __restrict__ x, const float* __restrict__ scale,
    const float* __restrict__ bias, _Float16* __restrict__ out)
{
    __shared__ float red[256];
    const int row = blockIdx.x;
    const int tid = threadIdx.x;
    const float* xr = x + (size_t)row * DMODEL;
    const float a = xr[tid];
    const float b = xr[tid + 256];
    red[tid] = a + b;
    __syncthreads();
    for (int off = 128; off > 0; off >>= 1) {
        if (tid < off) red[tid] += red[tid + off];
        __syncthreads();
    }
    const float mu = red[0] * (1.0f / DMODEL);
    __syncthreads();
    const float da = a - mu, db = b - mu;
    red[tid] = da * da + db * db;
    __syncthreads();
    for (int off = 128; off > 0; off >>= 1) {
        if (tid < off) red[tid] += red[tid + off];
        __syncthreads();
    }
    const float rs = rsqrtf(red[0] * (1.0f / DMODEL) + 1e-5f);
    _Float16* o = out + (size_t)row * DMODEL;
    o[tid]       = (_Float16)(da * rs * scale[tid] + bias[tid]);
    o[tid + 256] = (_Float16)(db * rs * scale[tid + 256] + bias[tid + 256]);
}

__global__ void cvt_f32_f16(const float* __restrict__ in,
                            _Float16* __restrict__ out, int n)
{
    const int i = blockIdx.x * blockDim.x + threadIdx.x;
    if (i < n) out[i] = (_Float16)in[i];
}

// ---------------------------------------------------------------------------
extern "C" void kernel_launch(void* const* d_in, const int* in_sizes, int n_in,
                              void* d_out, int out_size, void* d_ws, size_t ws_size,
                              hipStream_t stream)
{
    (void)in_sizes; (void)n_in; (void)out_size; (void)ws_size;
    const float*         frames    = (const float*)d_in[0];
    const unsigned char* mask      = (const unsigned char*)d_in[1];
    const float*         W_in_proj = (const float*)d_in[2];
    const float*         b_in_proj = (const float*)d_in[3];
    const float*         pos_emb   = (const float*)d_in[4];
    const float*         ln_scale  = (const float*)d_in[5];
    const float*         ln_bias   = (const float*)d_in[6];
    const float*         W_in      = (const float*)d_in[7];
    const float*         b_in      = (const float*)d_in[8];
    const float*         W_out     = (const float*)d_in[9];
    const float*         b_out     = (const float*)d_in[10];
    const float*         fn_scale  = (const float*)d_in[11];
    const float*         fn_bias   = (const float*)d_in[12];
    const float*         W_frame   = (const float*)d_in[13];
    const float*         b_frame   = (const float*)d_in[14];
    const float*         W_sym     = (const float*)d_in[15];
    const float*         b_sym     = (const float*)d_in[16];
    float* out = (float*)d_out;

    // workspace carve-up (256B aligned slices)
    char* ws = (char*)d_ws;
    size_t off = 0;
    auto carve = [&](size_t bytes) -> void* {
        void* p = ws + off;
        off = (off + bytes + 255) & ~(size_t)255;
        return p;
    };
    _Float16* frames16 = (_Float16*)carve((size_t)BSROWS * NFRAME * 2);
    _Float16* Wip16    = (_Float16*)carve((size_t)NFRAME * DMODEL * 2);
    _Float16* Win16    = (_Float16*)carve((size_t)NLAYERS * DMODEL * 2 * DINNER * 2);
    _Float16* Wout16   = (_Float16*)carve((size_t)NLAYERS * DINNER * DMODEL * 2);
    _Float16* Wfr16    = (_Float16*)carve((size_t)DMODEL * NFRAME * 2);
    _Float16* Wsym16   = (_Float16*)carve((size_t)DMODEL * NVOCAB * 2);
    float*    x        = (float*)   carve((size_t)BSROWS * DMODEL * 4);
    _Float16* norm16   = (_Float16*)carve((size_t)BSROWS * DMODEL * 2);
    _Float16* act16    = (_Float16*)carve((size_t)BSROWS * DINNER * 2);

    auto cvt = [&](const float* s, _Float16* d, int n) {
        cvt_f32_f16<<<(n + 255) / 256, 256, 0, stream>>>(s, d, n);
    };
    cvt(frames,    frames16, BSROWS * NFRAME);
    cvt(W_in_proj, Wip16,    NFRAME * DMODEL);
    cvt(W_in,      Win16,    NLAYERS * DMODEL * 2 * DINNER);
    cvt(W_out,     Wout16,   NLAYERS * DINNER * DMODEL);
    cvt(W_frame,   Wfr16,    DMODEL * NFRAME);
    cvt(W_sym,     Wsym16,   DMODEL * NVOCAB);

    const dim3 blk(256);
    // x = frames @ W_in_proj + b + pos_emb
    gemm_wmma_f16<EPI_INPROJ><<<dim3(BSROWS / 128, DMODEL / 64), blk, 0, stream>>>(
        frames16, Wip16, b_in_proj, x, nullptr, pos_emb, nullptr,
        NFRAME, DMODEL, DMODEL, 0);

    for (int l = 0; l < NLAYERS; ++l) {
        layernorm_f16<<<BSROWS, 256, 0, stream>>>(
            x, ln_scale + (size_t)l * DMODEL, ln_bias + (size_t)l * DMODEL, norm16);
        // act = u * silu(g),  [u|g] = norm @ W_in[l] + b_in[l]
        gemm_wmma_f16<EPI_GLU><<<dim3(BSROWS / 128, DINNER / 64), blk, 0, stream>>>(
            norm16, Win16 + (size_t)l * DMODEL * 2 * DINNER,
            b_in + (size_t)l * 2 * DINNER,
            nullptr, act16, nullptr, nullptr,
            DMODEL, 2 * DINNER, DINNER, DINNER);
        // x = (x + act @ W_out[l] + b_out[l]) * decay(s)
        gemm_wmma_f16<EPI_OUT><<<dim3(BSROWS / 128, DMODEL / 64), blk, 0, stream>>>(
            act16, Wout16 + (size_t)l * DINNER * DMODEL,
            b_out + (size_t)l * DMODEL,
            x, nullptr, nullptr, nullptr,
            DINNER, DMODEL, DMODEL, 0);
    }

    layernorm_f16<<<BSROWS, 256, 0, stream>>>(x, fn_scale, fn_bias, norm16);
    // heads (masked)
    gemm_wmma_f16<EPI_HEAD><<<dim3(BSROWS / 128, NFRAME / 64), blk, 0, stream>>>(
        norm16, Wfr16, b_frame, out, nullptr, nullptr, mask,
        DMODEL, NFRAME, NFRAME, 0);
    gemm_wmma_f16<EPI_HEAD><<<dim3(BSROWS / 128, NVOCAB / 64), blk, 0, stream>>>(
        norm16, Wsym16, b_sym, out + (size_t)BSROWS * NFRAME, nullptr, nullptr, mask,
        DMODEL, NVOCAB, NVOCAB, 0);
}
